// KNeighborsVC_38620345926213
// MI455X (gfx1250) — matrix operand order repaired and checked by reading, not compiled, not run on hardware
//
#include <hip/hip_runtime.h>
#include <hip/hip_bf16.h>
#include <cfloat>

// ---------------- problem constants ----------------
#define TQ    2000
#define NM    100000
#define NMPAD 100800               // padded rows so edge tiles stay in-buffer
#define DDIM  1024
#define TOPK  4

// ---------------- tiling ----------------
#define M_TILE 128                 // rows per WG (8 waves x 16)
#define N_SUB  64                  // cols per chunk step
#define N_ITER 25                  // chunk steps per WG -> 1600 cols
#define NCB    63                  // ceil(100000 / 1600)
#define QROWS  2048                // padded query rows in ws

#define K_STEP 64                  // k per B-tile (2 WMMA k-halves)
#define BROW   72                  // bf16 per B row: 64 data + 8 pad (144 B)
#define SIMROW 68                  // f32 per sim row: 64 data + 4 pad
#define SIMWAVE (16 * SIMROW)

typedef __attribute__((ext_vector_type(4)))  __bf16 v4bf;
typedef __attribute__((ext_vector_type(8)))  __bf16 v8bf;
typedef __attribute__((ext_vector_type(16))) __bf16 v16bf;
typedef __attribute__((ext_vector_type(8)))  float  v8f;
typedef __attribute__((ext_vector_type(4)))  unsigned int u32x4;
typedef __attribute__((ext_vector_type(4)))  int  i32x4;
typedef __attribute__((ext_vector_type(8)))  int  i32x8;

#define CAT16(lo, hi) \
  __builtin_shufflevector((lo), (hi), 0,1,2,3,4,5,6,7,8,9,10,11,12,13,14,15)

#ifndef __has_builtin
#define __has_builtin(x) 0
#endif
#if __has_builtin(__builtin_amdgcn_tensor_load_to_lds)
#define HAVE_TDM 1
#else
#define HAVE_TDM 0
#endif

// ---------------------------------------------------------------
// TDM: DMA a 64-row x 64-bf16 tile of mn (row stride 1024 elems) into LDS,
// padding each 128B row with 16B so LDS row stride is 144B (bank-safe).
// D# packing per CDNA5 ISA 8.3/8.4 (2D tile, groups 2/3 zero).
// ---------------------------------------------------------------
#if HAVE_TDM
__device__ __forceinline__
void tdm_load_b(unsigned lds_addr, unsigned long long ga) {
  u32x4 g0;
  g0[0] = 1u;                                   // count=1, user descriptor
  g0[1] = lds_addr;                             // LDS byte address
  g0[2] = (unsigned)ga;                         // global_addr[31:0]
  g0[3] = (unsigned)((ga >> 32) & 0x01FFFFFFu)  // global_addr[56:32]
        | 0x80000000u;                          // type = 2 ("image")
  i32x8 g1;
  g1[0] = (int)((1u << 16)      // data_size = 1 (2 bytes)
        |       (1u << 20)      // pad_enable
        |       (4u << 22)      // pad_interval: 32 dwords (128 B) per pad
        |       (3u << 25));    // pad_amount: 4 dwords (16 B)
  g1[1] = (int)(1024u << 16);                   // tensor_dim0 = 1024 (lo16)
  g1[2] = (int)(0x89C0u << 16);                 // dim0 hi16=0 | tensor_dim1 lo16 (100800)
  g1[3] = (int)(1u | (64u << 16));              // tensor_dim1 hi16=1 | tile_dim0 = 64
  g1[4] = 64;                                   // tile_dim1 = 64, tile_dim2 = 0
  g1[5] = 1024;                                 // tensor_dim0_stride lo32 = 1024
  g1[6] = 0;
  g1[7] = 0;
  i32x4 gz = {0, 0, 0, 0};
#if defined(__clang_major__) && (__clang_major__ >= 23)
  i32x8 gz8 = {0, 0, 0, 0, 0, 0, 0, 0};
  __builtin_amdgcn_tensor_load_to_lds(g0, g1, gz, gz, gz8, 0);
#else
  __builtin_amdgcn_tensor_load_to_lds(g0, g1, gz, gz, 0);
#endif
}
#endif

// ---------------------------------------------------------------
// Kernel 1: mn[row] = bf16( matching_set[row] / max(||row||, 1e-12) )
// One wave per row (row cached in 32 regs/lane), 8 rows per block.
// ---------------------------------------------------------------
__global__ __launch_bounds__(256)
void mnorm_cast_kernel(const float* __restrict__ m, __bf16* __restrict__ mn) {
  const int wave = threadIdx.x >> 5;
  const int lane = threadIdx.x & 31;
  const int row  = blockIdx.x * 8 + wave;
  if (row >= NM) return;
  const float* p = m + (size_t)row * DDIM;
  float v[32];
  float ss = 0.f;
  #pragma unroll
  for (int j = 0; j < 32; ++j) { v[j] = p[lane + 32 * j]; ss += v[j] * v[j]; }
  #pragma unroll
  for (int off = 16; off > 0; off >>= 1) ss += __shfl_xor(ss, off, 32);
  const float inv = 1.0f / fmaxf(sqrtf(ss), 1e-12f);
  __bf16* q = mn + (size_t)row * DDIM;
  #pragma unroll
  for (int j = 0; j < 32; ++j) q[lane + 32 * j] = (__bf16)(v[j] * inv);
}

// ---------------------------------------------------------------
// Kernel 2: cast queries fp32 -> bf16 (norms skipped: per-row positive
// scale never changes a row's top-k).
// ---------------------------------------------------------------
__global__ __launch_bounds__(256)
void qcast_kernel(const float* __restrict__ q, __bf16* __restrict__ qn) {
  const int i = (blockIdx.x * 256 + threadIdx.x) * 4;
  if (i >= TQ * DDIM) return;
  float4 v = *(const float4*)(q + i);
  v4bf o;
  o[0] = (__bf16)v.x; o[1] = (__bf16)v.y; o[2] = (__bf16)v.z; o[3] = (__bf16)v.w;
  *(v4bf*)(qn + i) = o;
}

// ---------------------------------------------------------------
// Kernel 3: bf16-WMMA GEMM + fused per-row running top-4.
// B tiles arrive via double-buffered TDM DMA (TENSORcnt), overlapping
// the WMMA pipeline; sims dumped to LDS, scanned by lanes 0-15.
// ---------------------------------------------------------------
#define TOPK_SWAP(a, b)                                              \
  if (tv[a] > tv[b]) {                                               \
    float tf = tv[a]; tv[a] = tv[b]; tv[b] = tf;                     \
    int   tn = ti[a]; ti[a] = ti[b]; ti[b] = tn;                     \
  }

__global__ __launch_bounds__(256)
void gemm_topk_kernel(const __bf16* __restrict__ mn,
                      const __bf16* __restrict__ qn,
                      float* __restrict__ pval,
                      int*   __restrict__ pidx) {
  __shared__ __align__(16) __bf16 b_lds[2][64 * BROW];
  __shared__ float sim_lds[8 * SIMWAVE];

  const int tid   = threadIdx.x;
  const int wave  = tid >> 5;
  const int lane  = tid & 31;
  const int l15   = lane & 15;
  const int hi    = lane >> 4;
  const int mbase = blockIdx.y * M_TILE;
  const int myrow = mbase + wave * 16 + l15;          // < QROWS by design

#if HAVE_TDM
  const unsigned lds0 = (unsigned)(size_t)&b_lds[0][0];
  const unsigned lds1 = (unsigned)(size_t)&b_lds[1][0];
  const unsigned long long mnb = (unsigned long long)(size_t)mn;
#endif

  float tv[TOPK]; int ti[TOPK];
  #pragma unroll
  for (int j = 0; j < TOPK; ++j) { tv[j] = -FLT_MAX; ti[j] = 0; }

  for (int iter = 0; iter < N_ITER; ++iter) {
    const int nbase = (blockIdx.x * N_ITER + iter) * N_SUB;

    v8f acc[4];
    #pragma unroll
    for (int t = 0; t < 4; ++t) acc[t] = (v8f){0.f,0.f,0.f,0.f,0.f,0.f,0.f,0.f};

#if HAVE_TDM
    if (wave == 0)                                    // prologue DMA, buf 0
      tdm_load_b(lds0, mnb + (size_t)nbase * DDIM * 2);
#endif

    for (int k0 = 0; k0 < DDIM; k0 += K_STEP) {
      const int cur = (k0 >> 6) & 1;
      __syncthreads();              // all reads of buf[cur^1] (iter k0-64) done
#if HAVE_TDM
      if (wave == 0) {
        if (k0 + K_STEP < DDIM) {   // DMA next tile, then wait current done
          tdm_load_b(cur ? lds0 : lds1,
                     mnb + ((size_t)nbase * DDIM + (size_t)(k0 + K_STEP)) * 2);
          __builtin_amdgcn_s_wait_tensorcnt(1);
        } else {
          __builtin_amdgcn_s_wait_tensorcnt(0);
        }
      }
#else
      { // fallback: cooperative copy of the bf16 tile
        const int rowc = tid >> 2;
        const int offc = (tid & 3) * 16;
        const __bf16* src = mn + (size_t)(nbase + rowc) * DDIM + k0 + offc;
        __bf16* dst = &b_lds[cur][rowc * BROW + offc];
        *(v8bf*)dst       = *(const v8bf*)src;
        *(v8bf*)(dst + 8) = *(const v8bf*)(src + 8);
      }
#endif
      __syncthreads();              // publish buf[cur]

      // A operand: 16x32 bf16 x2 (ISA 7.12.2: lanes 0-15 k{0-7,16-23},
      // lanes 16-31 k{8-15,24-31})
      const __bf16* ap = qn + (size_t)myrow * DDIM + k0 + hi * 8;
      v16bf A0 = CAT16(*(const v8bf*)ap,        *(const v8bf*)(ap + 16));
      v16bf A1 = CAT16(*(const v8bf*)(ap + 32), *(const v8bf*)(ap + 48));
      if (k0 + K_STEP < DDIM) __builtin_prefetch(ap + K_STEP, 0, 1);

      // B operand: 32x16 bf16 per subtile; lanes 0-15 k{0-7,16-23}, 16-31 k{8-15,24-31}
      #pragma unroll
      for (int t = 0; t < 4; ++t) {
        const __bf16* bp = &b_lds[cur][(t * 16 + l15) * BROW + hi * 8];
        v16bf B0 = CAT16(*(const v8bf*)bp,        *(const v8bf*)(bp + 16));
        v16bf B1 = CAT16(*(const v8bf*)(bp + 32), *(const v8bf*)(bp + 48));
        acc[t] = __builtin_amdgcn_wmma_f32_16x16x32_bf16(
            false, A0, false, B0, (short)0, acc[t], false, false);
        acc[t] = __builtin_amdgcn_wmma_f32_16x16x32_bf16(
            false, A1, false, B1, (short)0, acc[t], false, false);
      }
    }

    // Dump this wave's 16x64 sims (C layout: VGPR r -> row r or r+8)
    float* sw = sim_lds + wave * SIMWAVE;
    #pragma unroll
    for (int t = 0; t < 4; ++t) {
      #pragma unroll
      for (int r = 0; r < 8; ++r)
        sw[(r + hi * 8) * SIMROW + t * 16 + l15] = acc[t][r];
    }

    // Lanes 0-15: scan one row each; mn pre-normalized so sims are final.
    if (lane < 16) {
      const float* rowp = sw + l15 * SIMROW;
      #pragma unroll 4
      for (int c = 0; c < N_SUB; ++c) {
        const int n = nbase + c;
        float v = (n < NM) ? rowp[c] : -FLT_MAX;
        if (v > tv[3]) {
          tv[3] = v; ti[3] = n;
          TOPK_SWAP(3, 2); TOPK_SWAP(2, 1); TOPK_SWAP(1, 0);
        }
      }
    }
  }

  if (lane < 16) {
    const int    row  = mbase + wave * 16 + l15;
    const size_t base = ((size_t)row * NCB + blockIdx.x) * TOPK;
    #pragma unroll
    for (int j = 0; j < TOPK; ++j) { pval[base + j] = tv[j]; pidx[base + j] = ti[j]; }
  }
}

// ---------------------------------------------------------------
// Kernel 4: merge NCB*4 partials per row -> top-4 -> mean of synth rows
// ---------------------------------------------------------------
__global__ __launch_bounds__(256)
void topk_gather_kernel(const float* __restrict__ pval,
                        const int*   __restrict__ pidx,
                        const float* __restrict__ synth,
                        float*       __restrict__ out) {
  __shared__ int s_idx[TOPK];
  const int r = blockIdx.x;

  if (threadIdx.x == 0) {
    float tv[TOPK]; int ti[TOPK];
    #pragma unroll
    for (int j = 0; j < TOPK; ++j) { tv[j] = -FLT_MAX; ti[j] = 0; }
    const size_t base = (size_t)r * NCB * TOPK;
    for (int c = 0; c < NCB * TOPK; ++c) {
      float v = pval[base + c];
      int   n = pidx[base + c];
      if (v > tv[3]) {
        tv[3] = v; ti[3] = n;
        TOPK_SWAP(3, 2); TOPK_SWAP(2, 1); TOPK_SWAP(1, 0);
      }
    }
    #pragma unroll
    for (int j = 0; j < TOPK; ++j) s_idx[j] = ti[j];
  }
  __syncthreads();

  const size_t i0 = (size_t)s_idx[0] * DDIM;
  const size_t i1 = (size_t)s_idx[1] * DDIM;
  const size_t i2 = (size_t)s_idx[2] * DDIM;
  const size_t i3 = (size_t)s_idx[3] * DDIM;
  for (int d = threadIdx.x; d < DDIM; d += blockDim.x) {
    float s = synth[i0 + d] + synth[i1 + d] + synth[i2 + d] + synth[i3 + d];
    out[(size_t)r * DDIM + d] = 0.25f * s;
  }
}

// ---------------------------------------------------------------
extern "C" void kernel_launch(void* const* d_in, const int* in_sizes, int n_in,
                              void* d_out, int out_size, void* d_ws, size_t ws_size,
                              hipStream_t stream) {
  (void)in_sizes; (void)n_in; (void)out_size; (void)ws_size;

  const float* q     = (const float*)d_in[0];   // (2000, 1024)
  const float* match = (const float*)d_in[1];   // (100000, 1024)
  const float* synth = (const float*)d_in[2];   // (100000, 1024)
  float* out = (float*)d_out;

  // workspace layout (~215 MB):
  //   qn: 2048x1024 bf16 (4 MB) | mn: 100800x1024 bf16 (206.4 MB)
  //   pval/pidx: 2048*63*4 each (2 MB + 2 MB)
  char* ws = (char*)d_ws;
  const size_t QN_BYTES   = (size_t)QROWS * DDIM * 2;
  const size_t MN_BYTES   = (size_t)NMPAD * DDIM * 2;
  const size_t PART_ELEMS = (size_t)QROWS * NCB * TOPK;

  __bf16* qn   = (__bf16*)ws;
  __bf16* mn   = (__bf16*)(ws + QN_BYTES);
  float*  pval = (float*)(ws + QN_BYTES + MN_BYTES);
  int*    pidx = (int*)  (ws + QN_BYTES + MN_BYTES + PART_ELEMS * 4);

  mnorm_cast_kernel<<<dim3(NM / 8), dim3(256), 0, stream>>>(match, mn);
  qcast_kernel<<<dim3((TQ * DDIM) / (256 * 4)), dim3(256), 0, stream>>>(q, qn);
  gemm_topk_kernel<<<dim3(NCB, (TQ + M_TILE - 1) / M_TILE), dim3(256), 0, stream>>>(
      mn, qn, pval, pidx);
  topk_gather_kernel<<<dim3(TQ), dim3(256), 0, stream>>>(pval, pidx, synth, out);
}
